// MoEMambaBlock_88699664597232
// MI455X (gfx1250) — matrix operand order repaired and checked by reading, not compiled, not run on hardware
//
#include <hip/hip_runtime.h>
#include <hip/hip_bf16.h>
#include <math.h>
#include <stdint.h>

// ---------------------------------------------------------------------------
// MoE-Mamba block for MI455X (gfx1250, wave32, WMMA, LDS-tiled GEMMs with
// async global->LDS staging for the A tile).
// Pipeline: gemm_in (tiled WMMA bf16) -> conv+silu -> gemm_x (small WMMA) ->
//           gemm_dt (tiled WMMA, fused softplus) -> selective scan ->
//           gemm_out (tiled WMMA, fused residual) -> layernorm + leakyrelu
// ---------------------------------------------------------------------------

typedef __bf16 v16bf __attribute__((ext_vector_type(16)));
typedef __bf16 v8bf  __attribute__((ext_vector_type(8)));
typedef float  v8f   __attribute__((ext_vector_type(8)));
typedef float  f32x4 __attribute__((ext_vector_type(4)));

#define D_MODEL 1024
#define D_INNER 2048
#define D_STATE 16
#define DT_RANK 64
#define NROWS   4096   // BATCH * SEQ
#define SEQ_LEN 1024

// Tiled GEMM geometry
#define BM 128
#define BN 128
#define KT 32
#define APADF 4        // pad A rows by 4 f32  (row stride 144B, 16B-aligned)
#define BPADH 8        // pad B rows by 8 bf16 (row stride 80B,  16B-aligned)

// Workspace layout (float offsets). xz region (16.78M floats) is reused for
// dt and y_gated once the conv kernel has consumed xz.
static const size_t OFF_DT   = 0;                    // 4096*2048
static const size_t OFF_YG   = 8388608;              // 4096*2048
static const size_t OFF_XC   = 16777216;             // 4096*2048 xc after conv+silu
static const size_t OFF_ZS   = 25165824;             // 4096*2048 silu(z)
static const size_t OFF_XDBL = 33554432;             // 4096*96
static const size_t OFF_H    = 33947648;             // 4096*1024 pre-LN hidden

__device__ __forceinline__ float silu_f(float v) { return v / (1.f + __expf(-v)); }

__device__ __forceinline__ v16bf join8(v8bf lo, v8bf hi) {
    return __builtin_shufflevector(lo, hi, 0, 1, 2, 3, 4, 5, 6, 7,
                                           8, 9, 10, 11, 12, 13, 14, 15);
}

// Async copy of 16 bytes global -> LDS (per-lane addresses), ASYNCcnt-tracked.
__device__ __forceinline__ void async_g2l_b128(uint32_t lds_addr, const void* gptr) {
    const uint64_t ga = (uint64_t)(uintptr_t)gptr;
    asm volatile("global_load_async_to_lds_b128 %0, %1, off"
                 :: "v"(lds_addr), "v"(ga) : "memory");
}
__device__ __forceinline__ void wait_async0() {
#if __has_builtin(__builtin_amdgcn_s_wait_asynccnt)
    __builtin_amdgcn_s_wait_asynccnt(0);
#else
    asm volatile("s_wait_asynccnt 0x0" ::: "memory");
#endif
}

// ---------------------------------------------------------------------------
// LDS-tiled bf16-WMMA GEMM: C[M,N] = epilogue(A[M,K] @ B[K,N]), row-major f32.
// Block = 256 threads (8 waves) computes a 128x128 tile; waves arranged
// 4(M) x 2(N); each wave owns 32x64 = 2x4 WMMA accumulators.
// A tile staged f32 via global_load_async_to_lds_b128 (converted to bf16 at
// fragment build); B tile staged transposed as bf16 via VALU cvt + ds_store.
// MODE 0: plain store. MODE 1: softplus(acc + bias[col]). MODE 2: acc + E[row,col].
// Requires M%128==0, N%128==0, K%32==0.
// ---------------------------------------------------------------------------
template <int MODE>
__global__ __launch_bounds__(256)
void gemm_wmma_tiled(const float* __restrict__ A, const float* __restrict__ B,
                     float* __restrict__ C, const float* __restrict__ E,
                     int K, int lda, int ldb, int ldc)
{
    __shared__ __align__(16) float  As[BM][KT + APADF];   // f32 [row][k]
    __shared__ __align__(16) __bf16 Bs[BN][KT + BPADH];   // bf16 transposed [col][k]

    const int tid   = threadIdx.x;
    const int lane  = tid & 31;
    const int w     = tid >> 5;            // wave id 0..7
    const int wm    = w >> 1;              // 0..3 -> 32-row group
    const int wn    = w & 1;               // 0..1 -> 64-col group
    const int row15 = lane & 15;
    const int hi    = lane >> 4;

    const int rowBase = blockIdx.y * BM;
    const int colBase = blockIdx.x * BN;

    v8f acc[2][4];
#pragma unroll
    for (int mi = 0; mi < 2; ++mi)
#pragma unroll
        for (int ni = 0; ni < 4; ++ni) acc[mi][ni] = (v8f){};

    for (int k0 = 0; k0 < K; k0 += KT) {
        __syncthreads();   // previous fragments fully consumed

        // ---- A tile: 128x32 f32, async global -> LDS, 4x b128 per thread ----
#pragma unroll
        for (int i = 0; i < 4; ++i) {
            const int e  = tid + 256 * i;          // 16B chunk id, 0..1023
            const int ar = e >> 3;                 // row 0..127
            const int ak = (e & 7) * 4;            // k: 0,4,...,28
            async_g2l_b128((uint32_t)(uintptr_t)&As[ar][ak],
                           &A[(size_t)(rowBase + ar) * lda + (k0 + ak)]);
        }

        // ---- B tile: 32x128 -> transposed bf16 [col][k] ----
#pragma unroll
        for (int i = 0; i < 16; ++i) {
            const int e  = tid + 256 * i;
            const int bk = e >> 7, bn = e & 127;   // B: k, col
            Bs[bn][bk] = (__bf16)B[(size_t)(k0 + bk) * ldb + (colBase + bn)];
        }
        if (k0 + KT < K)   // hint next B K-tile into cache (global_prefetch_b8)
            __builtin_prefetch(&B[(size_t)(k0 + KT + (tid >> 4)) * ldb + colBase + (tid & 15) * 8], 0, 0);

        wait_async0();     // own wave's async copies landed in LDS
        __syncthreads();   // everyone's staging visible

        // ---- A fragments: 4x ds_load_b128 f32 + cvt to bf16 per subtile ----
        v16bf afrag[2];
#pragma unroll
        for (int mi = 0; mi < 2; ++mi) {
            const int r = wm * 32 + mi * 16 + row15;
            f32x4 a0 = *(const f32x4*)&As[r][hi * 8];
            f32x4 a1 = *(const f32x4*)&As[r][hi * 8 + 4];
            f32x4 a2 = *(const f32x4*)&As[r][16 + hi * 8];
            f32x4 a3 = *(const f32x4*)&As[r][16 + hi * 8 + 4];
            v16bf af;
#pragma unroll
            for (int j = 0; j < 4; ++j) {
                af[j]      = (__bf16)a0[j];
                af[4 + j]  = (__bf16)a1[j];
                af[8 + j]  = (__bf16)a2[j];
                af[12 + j] = (__bf16)a3[j];
            }
            afrag[mi] = af;
        }
        // ---- B fragments: two 16B bf16 runs each -> ds_load_b128 ----
        v16bf bfrag[4];
#pragma unroll
        for (int ni = 0; ni < 4; ++ni) {
            const int c = wn * 64 + ni * 16 + row15;
            v8bf lo = *(const v8bf*)&Bs[c][hi * 8];
            v8bf hh = *(const v8bf*)&Bs[c][16 + hi * 8];
            bfrag[ni] = join8(lo, hh);
        }
#pragma unroll
        for (int mi = 0; mi < 2; ++mi)
#pragma unroll
            for (int ni = 0; ni < 4; ++ni)
                acc[mi][ni] = __builtin_amdgcn_wmma_f32_16x16x32_bf16(
                    false, afrag[mi], false, bfrag[ni],
                    (short)0, acc[mi][ni], false, false);
    }

    // ---- epilogue: f32 C/D layout (lane col = lane&15, vgpr r -> row) ----
#pragma unroll
    for (int mi = 0; mi < 2; ++mi) {
#pragma unroll
        for (int ni = 0; ni < 4; ++ni) {
            const int gN = colBase + wn * 64 + ni * 16 + row15;
#pragma unroll
            for (int r = 0; r < 8; ++r) {
                const int gM = rowBase + wm * 32 + mi * 16 + hi * 8 + r;
                float v = acc[mi][ni][r];
                if (MODE == 1) {            // dt proj: softplus(acc + b_dt)
                    v += E[gN];
                    v = (v > 20.f) ? v : log1pf(__expf(v));
                } else if (MODE == 2) {     // out proj: + residual x
                    v += E[(size_t)gM * ldc + gN];
                }
                C[(size_t)gM * ldc + gN] = v;
            }
        }
    }
}

// ---------------------------------------------------------------------------
// Small-N WMMA GEMM (one wave per 16x16 tile) for N not divisible by 128.
// Used for x_dbl = xc @ W_x (N=96).
// ---------------------------------------------------------------------------
__global__ __launch_bounds__(32)
void gemm_wmma_small(const float* __restrict__ A, const float* __restrict__ B,
                     float* __restrict__ C, int K, int lda, int ldb, int ldc)
{
    const int lane  = threadIdx.x & 31;
    const int row15 = lane & 15;
    const int hi    = lane >> 4;
    const int aRow  = blockIdx.y * 16 + row15;
    const int bCol  = blockIdx.x * 16 + row15;

    v8f acc = {};
    for (int k0 = 0; k0 < K; k0 += 32) {
        v16bf af, bf;
#pragma unroll
        for (int i = 0; i < 16; ++i) {
            const int v = i >> 1, p = i & 1;
            const int ka = (v < 4) ? (hi * 8 + 2 * v + p)
                                   : (16 + hi * 8 + 2 * (v - 4) + p);
            af[i] = (__bf16)A[(size_t)aRow * lda + (k0 + ka)];
            const int kb = (i < 8) ? (hi * 8 + i) : (16 + hi * 8 + (i - 8));
            bf[i] = (__bf16)B[(size_t)(k0 + kb) * ldb + bCol];
        }
        acc = __builtin_amdgcn_wmma_f32_16x16x32_bf16(
            false, af, false, bf, (short)0, acc, false, false);
    }
#pragma unroll
    for (int r = 0; r < 8; ++r) {
        const int gM = blockIdx.y * 16 + hi * 8 + r;
        C[(size_t)gM * ldc + bCol] = acc[r];
    }
}

// ---------------------------------------------------------------------------
// Depthwise causal conv (D_CONV=4) + bias + SiLU on xc; SiLU on z.
// ---------------------------------------------------------------------------
__global__ void conv_silu_kernel(const float* __restrict__ xz,
                                 const float* __restrict__ conv_w,
                                 const float* __restrict__ conv_b,
                                 float* __restrict__ xc_act,
                                 float* __restrict__ zs)
{
    const int idx = blockIdx.x * blockDim.x + threadIdx.x;   // 0 .. 4096*2048-1
    const int d   = idx & (D_INNER - 1);
    const int row = idx >> 11;                               // b*SEQ + t
    const int t   = row & (SEQ_LEN - 1);

    float s = conv_b[d];
#pragma unroll
    for (int j = 0; j < 4; ++j) {
        const int tt = t - 3 + j;
        if (tt >= 0)
            s += conv_w[d * 4 + j] * xz[(size_t)(row - 3 + j) * 4096 + d];
    }
    xc_act[idx] = silu_f(s);
    zs[idx]     = silu_f(xz[(size_t)row * 4096 + D_INNER + d]);
}

// ---------------------------------------------------------------------------
// Selective scan: one lane per (b, d) row; 16-state vector in registers.
// ---------------------------------------------------------------------------
__global__ void scan_kernel(const float* __restrict__ dtb,
                            const float* __restrict__ xdbl,
                            const float* __restrict__ u_buf,
                            const float* __restrict__ zs,
                            const float* __restrict__ A_log,
                            const float* __restrict__ Dvec,
                            float* __restrict__ y_g)
{
    const int gid = blockIdx.x * blockDim.x + threadIdx.x;   // 0..8191
    const int d   = gid & (D_INNER - 1);
    const int b   = gid >> 11;

    float Ad[D_STATE], h[D_STATE];
#pragma unroll
    for (int n = 0; n < D_STATE; ++n) {
        Ad[n] = -__expf(A_log[d * D_STATE + n]);
        h[n]  = 0.f;
    }
    const float Dd = Dvec[d];

    for (int t = 0; t < SEQ_LEN; ++t) {
        const int r = (b << 10) + t;
        const float dtv = dtb[(size_t)r * D_INNER + d];
        const float u   = u_buf[(size_t)r * D_INNER + d];
        const float* Bp = xdbl + (size_t)r * 96 + DT_RANK;
        const float* Cp = Bp + D_STATE;
        float y = 0.f;
#pragma unroll
        for (int n = 0; n < D_STATE; ++n) {
            h[n] = __expf(dtv * Ad[n]) * h[n] + dtv * Bp[n] * u;
            y += h[n] * Cp[n];
        }
        y += u * Dd;
        y_g[(size_t)r * D_INNER + d] = y * zs[(size_t)r * D_INNER + d];
    }
}

// ---------------------------------------------------------------------------
// LayerNorm + LeakyReLU, one workgroup (8 waves) per row.
// ---------------------------------------------------------------------------
__global__ void ln_lrelu_kernel(const float* __restrict__ h,
                                const float* __restrict__ gamma,
                                const float* __restrict__ beta,
                                float* __restrict__ out)
{
    __shared__ float red[256];
    const int row = blockIdx.x;
    const int tid = threadIdx.x;
    const float* hr = h + (size_t)row * D_MODEL;

    float s = 0.f;
    for (int c = tid; c < D_MODEL; c += 256) s += hr[c];
    red[tid] = s; __syncthreads();
    for (int off = 128; off > 0; off >>= 1) {
        if (tid < off) red[tid] += red[tid + off];
        __syncthreads();
    }
    const float mu = red[0] * (1.f / D_MODEL);
    __syncthreads();

    float v = 0.f;
    for (int c = tid; c < D_MODEL; c += 256) {
        const float dlt = hr[c] - mu;
        v += dlt * dlt;
    }
    red[tid] = v; __syncthreads();
    for (int off = 128; off > 0; off >>= 1) {
        if (tid < off) red[tid] += red[tid + off];
        __syncthreads();
    }
    const float rstd = rsqrtf(red[0] * (1.f / D_MODEL) + 1e-5f);

    for (int c = tid; c < D_MODEL; c += 256) {
        float o = (hr[c] - mu) * rstd * gamma[c] + beta[c];
        out[(size_t)row * D_MODEL + c] = (o >= 0.f) ? o : 0.01f * o;
    }
}

// ---------------------------------------------------------------------------
extern "C" void kernel_launch(void* const* d_in, const int* in_sizes, int n_in,
                              void* d_out, int out_size, void* d_ws, size_t ws_size,
                              hipStream_t stream)
{
    const float* x      = (const float*)d_in[0];   // (4,1024,1024)
    const float* W_in   = (const float*)d_in[1];   // (1024,4096)
    const float* conv_w = (const float*)d_in[2];   // (2048,4)
    const float* conv_b = (const float*)d_in[3];   // (2048,)
    const float* W_x    = (const float*)d_in[4];   // (2048,96)
    const float* W_dt   = (const float*)d_in[5];   // (64,2048)
    const float* b_dt   = (const float*)d_in[6];   // (2048,)
    const float* A_log  = (const float*)d_in[7];   // (2048,16)
    const float* Dvec   = (const float*)d_in[8];   // (2048,)
    const float* W_out  = (const float*)d_in[9];   // (2048,1024)
    const float* gamma  = (const float*)d_in[10];  // (1024,)
    const float* beta   = (const float*)d_in[11];  // (1024,)
    float* out = (float*)d_out;
    float* ws  = (float*)d_ws;

    float* xz    = ws;            // aliases dt / y_g (xz dead after conv)
    float* dt    = ws + OFF_DT;
    float* y_g   = ws + OFF_YG;
    float* xc    = ws + OFF_XC;
    float* zs    = ws + OFF_ZS;
    float* x_dbl = ws + OFF_XDBL;
    float* hbuf  = ws + OFF_H;

    // 1) xz = x @ W_in                     M=4096 K=1024 N=4096
    gemm_wmma_tiled<0><<<dim3(4096 / BN, NROWS / BM), 256, 0, stream>>>(
        x, W_in, xz, nullptr, /*K=*/1024, 1024, 4096, 4096);

    // 2) depthwise causal conv + SiLU on xc; SiLU on z
    conv_silu_kernel<<<(NROWS * D_INNER) / 256, 256, 0, stream>>>(
        xz, conv_w, conv_b, xc, zs);

    // 3) x_dbl = xc @ W_x                  M=4096 K=2048 N=96
    gemm_wmma_small<<<dim3(96 / 16, NROWS / 16), 32, 0, stream>>>(
        xc, W_x, x_dbl, /*K=*/2048, 2048, 96, 96);

    // 4) dt = softplus(x_dbl[:, :64] @ W_dt + b_dt)   M=4096 K=64 N=2048
    gemm_wmma_tiled<1><<<dim3(2048 / BN, NROWS / BM), 256, 0, stream>>>(
        x_dbl, W_dt, dt, b_dt, /*K=*/64, /*lda=*/96, 2048, 2048);

    // 5) selective scan + skip (u*D) + gating by silu(z)
    scan_kernel<<<(4 * D_INNER) / 256, 256, 0, stream>>>(
        dt, x_dbl, xc, zs, A_log, Dvec, y_g);

    // 6) h = x + y_g @ W_out               M=4096 K=2048 N=1024
    gemm_wmma_tiled<2><<<dim3(1024 / BN, NROWS / BM), 256, 0, stream>>>(
        y_g, W_out, hbuf, x, /*K=*/2048, 2048, 1024, 1024);

    // 7) LayerNorm + LeakyReLU -> d_out
    ln_lrelu_kernel<<<NROWS, 256, 0, stream>>>(hbuf, gamma, beta, out);

    (void)in_sizes; (void)n_in; (void)out_size; (void)ws_size;
}
